// RGAT_22067541967000
// MI455X (gfx1250) — matrix operand order repaired and checked by reading, not compiled, not run on hardware
//
#include <hip/hip_runtime.h>

// ---------------- problem constants ----------------
#define Nn 50000
#define Ee 800000
#define Rr 8
#define Hh 128
#define Cc 64

typedef __attribute__((ext_vector_type(16))) _Float16 v16h;
typedef __attribute__((ext_vector_type(8)))  float    v8f;

// fast f32 global atomic add (lowers to global_atomic_add_f32)
__device__ __forceinline__ void atomicAddF32(float* p, float v) {
    unsafeAtomicAdd(p, v);
}

// ---------------- WMMA fragment helpers (ISA 7.12.2 layouts) ----------------
// A fragment (16x32 f16): lane(lo,hi); element j -> K = kb + (j>>3)*16 + hi*8 + (j&7)
__device__ __forceinline__ v16h load_a_frag(const _Float16* At, int lo, int hi, int kb) {
    v16h a;
    const _Float16* ap = &At[lo * 128 + kb + hi * 8];
    reinterpret_cast<uint4*>(&a)[0] = *reinterpret_cast<const uint4*>(ap);
    reinterpret_cast<uint4*>(&a)[1] = *reinterpret_cast<const uint4*>(ap + 16);
    return a;
}
// B fragment (32x16 f16): N = lo, element j -> K = kb + (j>>3)*16 + hi*8 + (j&7)  (Bcol = column, contiguous K)
__device__ __forceinline__ v16h load_b_frag(const _Float16* Bcol, int hi, int kb) {
    v16h b;
    const _Float16* bp = Bcol + kb + hi * 8;
    reinterpret_cast<uint4*>(&b)[0] = *reinterpret_cast<const uint4*>(bp);
    reinterpret_cast<uint4*>(&b)[1] = *reinterpret_cast<const uint4*>(bp + 16);
    return b;
}
// full K=128 GEMM for one 16x16 tile: loads all B fragments, then 4 back-to-back WMMAs
__device__ __forceinline__ v8f gemm_k128(const v16h* a, const _Float16* Bcol, int hi) {
    v16h b0 = load_b_frag(Bcol, hi, 0);
    v16h b1 = load_b_frag(Bcol, hi, 32);
    v16h b2 = load_b_frag(Bcol, hi, 64);
    v16h b3 = load_b_frag(Bcol, hi, 96);
    v8f c = {};
    c = __builtin_amdgcn_wmma_f32_16x16x32_f16(false, a[0], false, b0, (short)0, c, false, false);
    c = __builtin_amdgcn_wmma_f32_16x16x32_f16(false, a[1], false, b1, (short)0, c, false, false);
    c = __builtin_amdgcn_wmma_f32_16x16x32_f16(false, a[2], false, b2, (short)0, c, false, false);
    c = __builtin_amdgcn_wmma_f32_16x16x32_f16(false, a[3], false, b3, (short)0, c, false, false);
    return c;
}

// ---------------- utility kernels ----------------
__global__ void fill_u32_kernel(unsigned* __restrict__ p, unsigned v, int n) {
    int i = blockIdx.x * 256 + threadIdx.x;
    if (i < n) p[i] = v;
}

__global__ void f32_to_f16_kernel(const float* __restrict__ src, _Float16* __restrict__ dst, int n) {
    int i = blockIdx.x * 256 + threadIdx.x;
    if (i < n) dst[i] = (_Float16)src[i];
}

// W [R,F,O] f32 row-major  ->  WT [R,O,F] f16 (column of W contiguous in K)
__global__ void w_to_f16T_kernel(const float* __restrict__ W, _Float16* __restrict__ WT,
                                 int R_, int F, int O) {
    int i = blockIdx.x * 256 + threadIdx.x;
    int n = R_ * F * O;
    if (i >= n) return;
    int r = i / (F * O);
    int rem = i - r * F * O;
    int f = rem / O;
    int o = rem - f * O;
    WT[((size_t)r * O + o) * F + f] = (_Float16)W[i];
}

// Precompute fused projection vectors:  qx[n,r] = x_n . (W_r q),  kx[n,r] = x_n . (W_r k).
// BqkT[col][f] f16, col 0..7 = W_r q, col 8..15 = W_r k  (one 128x16 WMMA B-tile)
__global__ void wqk_kernel(const float* __restrict__ W, const float* __restrict__ q,
                           const float* __restrict__ k, _Float16* __restrict__ BqkT) {
    int i = blockIdx.x * 256 + threadIdx.x;     // 16*128 outputs
    if (i >= 16 * Hh) return;
    int col = i >> 7;
    int f   = i & (Hh - 1);
    int r   = col & 7;
    const float* vec  = (col < 8) ? q : k;
    const float* Wrow = W + ((size_t)r * Hh + f) * Hh;   // W[r][f][:], contiguous
    float s = 0.f;
    for (int o = 0; o < Hh; ++o) s += Wrow[o] * vec[o];
    BqkT[(size_t)col * Hh + f] = (_Float16)s;
}

// ---------------- fused WMMA GEMM + q/k projections ----------------
// One block per 16-row tile; LDS A-tile loaded once, A fragments kept in VGPRs.
//   xr[n,r,:] = xh[n,:] @ W[r]   (8 relations, f16 out; wave w owns cols [16w,16w+16))
//   wave 0 additionally computes the 16x16 projection tile via BqkT -> qx, kx (f32)
__global__ __launch_bounds__(256) void wmma_xr_qk_kernel(const _Float16* __restrict__ xh,
                                                         const _Float16* __restrict__ WhT,
                                                         const _Float16* __restrict__ BqkT,
                                                         _Float16* __restrict__ xr,
                                                         float* __restrict__ qx,
                                                         float* __restrict__ kx) {
    __shared__ _Float16 At[16 * 128];
    const int t  = threadIdx.x;
    const int m0 = blockIdx.x * 16;
    {   // cooperative A-tile load: 256 threads x 8 halves (16B) = 16x128 halves
        int row = t >> 4, chunk = t & 15;
        *reinterpret_cast<uint4*>(&At[row * 128 + chunk * 8]) =
            *reinterpret_cast<const uint4*>(xh + (size_t)(m0 + row) * 128 + chunk * 8);
    }
    __syncthreads();

    const int lane = t & 31, wave = t >> 5;
    const int lo = lane & 15, hi = lane >> 4;
    const int n0 = wave * 16;

    // A fragments: invariant across relations -- load from LDS once
    v16h a[4];
    a[0] = load_a_frag(At, lo, hi, 0);
    a[1] = load_a_frag(At, lo, hi, 32);
    a[2] = load_a_frag(At, lo, hi, 64);
    a[3] = load_a_frag(At, lo, hi, 96);

    if (wave == 0) {    // wave-uniform branch; EXEC all-1s inside (WMMA-safe)
        v8f c = gemm_k128(a, BqkT + (size_t)lo * 128, hi);
        float* base = (lo < 8) ? qx : kx;       // col 0..7 -> qx, 8..15 -> kx
        int rcol = lo & 7;
#pragma unroll
        for (int i = 0; i < 8; i++) {           // C/D: M = i + 8*hi, N = lo
            base[(size_t)(m0 + i + hi * 8) * Rr + rcol] = c[i];
        }
    }

    for (int r = 0; r < Rr; ++r) {
        v8f c = gemm_k128(a, WhT + ((size_t)r * Hh + n0 + lo) * 128, hi);
#pragma unroll
        for (int i = 0; i < 8; i++) {
            int m = m0 + i + hi * 8;
            xr[((size_t)m * Rr + r) * Hh + n0 + lo] = (_Float16)c[i];
        }
    }
}

// ---------------- WMMA GEMM: logits = hh @ lin_w + lin_b (128 -> 64, out f32) ----------------
// grid Nn/16, block 128 (4 waves, one per 16-col tile)
__global__ __launch_bounds__(128) void wmma_logits_kernel(const _Float16* __restrict__ hh,
                                                          const _Float16* __restrict__ WlT,
                                                          const float* __restrict__ lb,
                                                          float* __restrict__ out) {
    __shared__ _Float16 At[16 * 128];
    const int t  = threadIdx.x;
    const int m0 = blockIdx.x * 16;
    {   // 128 threads x 16 halves
        int row = t >> 3, chunk = t & 7;
        const uint4* g = reinterpret_cast<const uint4*>(hh + (size_t)(m0 + row) * 128 + chunk * 16);
        uint4* s = reinterpret_cast<uint4*>(&At[row * 128 + chunk * 16]);
        s[0] = g[0];
        s[1] = g[1];
    }
    __syncthreads();

    const int lane = t & 31, wave = t >> 5;
    const int lo = lane & 15, hi = lane >> 4;
    const int n0 = wave * 16;
    v16h a[4];
    a[0] = load_a_frag(At, lo, hi, 0);
    a[1] = load_a_frag(At, lo, hi, 32);
    a[2] = load_a_frag(At, lo, hi, 64);
    a[3] = load_a_frag(At, lo, hi, 96);
    v8f c = gemm_k128(a, WlT + (size_t)(n0 + lo) * 128, hi);
    float bias = lb[n0 + lo];
#pragma unroll
    for (int i = 0; i < 8; i++) {
        int m = m0 + i + hi * 8;
        out[(size_t)m * Cc + n0 + lo] = c[i] + bias;
    }
}

// ---------------- edge score + segment max (ordered-uint atomicMax) ----------------
__global__ void alpha_kernel(const int* __restrict__ src, const int* __restrict__ dst,
                             const int* __restrict__ et,
                             const float* __restrict__ qx, const float* __restrict__ kx,
                             float* __restrict__ alpha, unsigned* __restrict__ mEnc) {
    int e = blockIdx.x * 256 + threadIdx.x;
    if (e >= Ee) return;
    int s = src[e], d = dst[e], r = et[e];
    float a = qx[(size_t)d * Rr + r] + kx[(size_t)s * Rr + r];
    a = (a >= 0.f) ? a : 0.2f * a;                    // leaky_relu, slope 0.2
    alpha[e] = a;
    unsigned bits = __float_as_uint(a);
    unsigned enc  = (bits & 0x80000000u) ? ~bits : (bits | 0x80000000u);
    atomicMax(mEnc + d, enc);
}

// ---------------- exp + segment sum ----------------
__global__ void expw_kernel(const int* __restrict__ dst, const float* __restrict__ alpha,
                            const unsigned* __restrict__ mEnc,
                            float* __restrict__ ew, float* __restrict__ denom) {
    int e = blockIdx.x * 256 + threadIdx.x;
    if (e >= Ee) return;
    int d = dst[e];
    unsigned u = mEnc[d];
    unsigned bits = (u & 0x80000000u) ? (u & 0x7fffffffu) : ~u;
    float m = __uint_as_float(bits);
    float w = __expf(alpha[e] - m);
    ew[e] = w;
    atomicAddF32(denom + d, w);
}

// ---------------- weighted message scatter-add: wave per edge ----------------
__global__ __launch_bounds__(256) void agg_kernel(const int* __restrict__ src,
                                                  const int* __restrict__ dst,
                                                  const int* __restrict__ et,
                                                  const float* __restrict__ ew,
                                                  const float* __restrict__ denom,
                                                  const _Float16* __restrict__ xr,
                                                  float* __restrict__ hout) {
    int e    = blockIdx.x * 8 + (threadIdx.x >> 5);
    int lane = threadIdx.x & 31;
    int s = src[e], d = dst[e], r = et[e];
    float w = ew[e] / denom[d];
    const _Float16* msg = xr + ((size_t)s * Rr + r) * Hh;
    float* orow = hout + (size_t)d * Hh;
#pragma unroll
    for (int i = 0; i < 4; i++) {
        int o = lane + 32 * i;
        atomicAddF32(orow + o, w * (float)msg[o]);
    }
}

// ---------------- fused bias + relu + f16 convert (writes next-layer features) ----------------
__global__ void bias_relu_f16_kernel(const float* __restrict__ hacc, const float* __restrict__ b,
                                     _Float16* __restrict__ xh, int n) {
    int i = blockIdx.x * 256 + threadIdx.x;
    if (i >= n) return;
    float v = hacc[i] + b[i & (Hh - 1)];
    xh[i] = (_Float16)(v > 0.f ? v : 0.f);
}

// ---------------- log_softmax over C=64, one wave per node, in place ----------------
__global__ __launch_bounds__(256) void lsm_kernel(float* __restrict__ out) {
    int n    = blockIdx.x * 8 + (threadIdx.x >> 5);
    int lane = threadIdx.x & 31;
    float* row = out + (size_t)n * Cc;
    float v0 = row[lane], v1 = row[lane + 32];
    float mx = fmaxf(v0, v1);
    for (int off = 16; off; off >>= 1) mx = fmaxf(mx, __shfl_xor(mx, off, 32));
    float se = __expf(v0 - mx) + __expf(v1 - mx);
    for (int off = 16; off; off >>= 1) se += __shfl_xor(se, off, 32);
    float ls = mx + __logf(se);
    row[lane]      = v0 - ls;
    row[lane + 32] = v1 - ls;
}

// ---------------- launch ----------------
extern "C" void kernel_launch(void* const* d_in, const int* in_sizes, int n_in,
                              void* d_out, int out_size, void* d_ws, size_t ws_size,
                              hipStream_t stream) {
    (void)in_sizes; (void)n_in; (void)out_size; (void)ws_size;
    const float* x     = (const float*)d_in[0];
    const int*   ei    = (const int*)d_in[1];
    const int*   et    = (const int*)d_in[2];
    const float* W1    = (const float*)d_in[3];
    const float* q1    = (const float*)d_in[4];
    const float* k1    = (const float*)d_in[5];
    const float* b1    = (const float*)d_in[6];
    const float* W2    = (const float*)d_in[7];
    const float* q2    = (const float*)d_in[8];
    const float* k2    = (const float*)d_in[9];
    const float* b2    = (const float*)d_in[10];
    const float* W3    = (const float*)d_in[11];
    const float* q3    = (const float*)d_in[12];
    const float* k3    = (const float*)d_in[13];
    const float* b3    = (const float*)d_in[14];
    const float* lin_w = (const float*)d_in[15];
    const float* lin_b = (const float*)d_in[16];
    const int* src  = ei;        // edge_index[0]
    const int* dstI = ei + Ee;   // edge_index[1]

    // carve workspace (~151 MB)
    char* p = (char*)d_ws;
    auto carve = [&](size_t bytes) -> char* {
        char* r = p;
        p += (bytes + 255) & ~((size_t)255);
        return r;
    };
    _Float16* xh   = (_Float16*)carve((size_t)Nn * Hh * 2);          // f16 node features (in/out)
    _Float16* WhT  = (_Float16*)carve((size_t)Rr * Hh * Hh * 2);     // f16 W^T per layer
    _Float16* BqkT = (_Float16*)carve((size_t)16 * Hh * 2);          // fused q/k projection tile
    _Float16* xr   = (_Float16*)carve((size_t)Nn * Rr * Hh * 2);     // per-relation transforms
    float*    qx   = (float*)carve((size_t)Nn * Rr * 4);
    float*    kx   = (float*)carve((size_t)Nn * Rr * 4);
    float*    alp  = (float*)carve((size_t)Ee * 4);
    float*    ew   = (float*)carve((size_t)Ee * 4);
    unsigned* mEnc = (unsigned*)carve((size_t)Nn * 4);
    float*    den  = (float*)carve((size_t)Nn * 4);
    float*    hAcc = (float*)carve((size_t)Nn * Hh * 4);             // f32 aggregation buffer
    _Float16* WlT  = (_Float16*)carve((size_t)Cc * Hh * 2);

    const unsigned ENC_NEG_INF = 0x007FFFFFu;   // encode(-inf)

    auto layer = [&](const float* W, const float* q, const float* k, const float* b) {
        w_to_f16T_kernel<<<(Rr * Hh * Hh + 255) / 256, 256, 0, stream>>>(W, WhT, Rr, Hh, Hh);
        wqk_kernel<<<(16 * Hh + 255) / 256, 256, 0, stream>>>(W, q, k, BqkT);
        fill_u32_kernel<<<(Nn + 255) / 256, 256, 0, stream>>>(mEnc, ENC_NEG_INF, Nn);
        fill_u32_kernel<<<(Nn + 255) / 256, 256, 0, stream>>>((unsigned*)den, 0u, Nn);
        fill_u32_kernel<<<(Nn * Hh + 255) / 256, 256, 0, stream>>>((unsigned*)hAcc, 0u, Nn * Hh);
        wmma_xr_qk_kernel<<<Nn / 16, 256, 0, stream>>>(xh, WhT, BqkT, xr, qx, kx);
        alpha_kernel<<<(Ee + 255) / 256, 256, 0, stream>>>(src, dstI, et, qx, kx, alp, mEnc);
        expw_kernel<<<(Ee + 255) / 256, 256, 0, stream>>>(dstI, alp, mEnc, ew, den);
        agg_kernel<<<Ee / 8, 256, 0, stream>>>(src, dstI, et, ew, den, xr, hAcc);
        bias_relu_f16_kernel<<<(Nn * Hh + 255) / 256, 256, 0, stream>>>(hAcc, b, xh, Nn * Hh);
    };

    // initial feature conversion, then three fused RGAT layers
    f32_to_f16_kernel<<<(Nn * Hh + 255) / 256, 256, 0, stream>>>(x, xh, Nn * Hh);
    layer(W1, q1, k1, b1);
    layer(W2, q2, k2, b2);
    layer(W3, q3, k3, b3);

    // final linear + log_softmax (xh holds relu(h3) in f16)
    w_to_f16T_kernel<<<(Hh * Cc + 255) / 256, 256, 0, stream>>>(lin_w, WlT, 1, Hh, Cc);
    wmma_logits_kernel<<<Nn / 16, 128, 0, stream>>>(xh, WlT, lin_b, (float*)d_out);
    lsm_kernel<<<Nn / 8, 256, 0, stream>>>((float*)d_out);
}
